// GNNClassifier_65197603553466
// MI455X (gfx1250) — matrix operand order repaired and checked by reading, not compiled, not run on hardware
//
#include <hip/hip_runtime.h>
#include <hip/hip_bf16.h>

#define NN 20000
#define EE 320000
#define GG 64
#define AF 128
#define BF 32
#define HH 256
#define NH 8
#define DH 32
#define LL 4
#define MH 1024
#define INV_SQRT_DH 0.17677669529663687f
#define LN_EPS 1e-5f

typedef __attribute__((ext_vector_type(16))) __bf16        v16bf;
typedef __attribute__((ext_vector_type(8)))  float         v8f;
typedef __attribute__((ext_vector_type(4)))  unsigned int  v4u;

union Frag {
    v16bf v;
    v4u   q[2];
};

__device__ __forceinline__ unsigned short f2bf(float f) {
    unsigned int u = __float_as_uint(f);
    unsigned int r = u + 0x7FFFu + ((u >> 16) & 1u);
    return (unsigned short)(r >> 16);
}

// ---------------------------------------------------------------------------
// Generic GEMM: C[M,Nc] = A[M,K](bf16) * Bt[Nc,K](bf16, pre-transposed) + bias
// One wave per 32x32 macro-tile: 2 A frags x 2 B frags -> 4 WMMA accumulators,
// halving per-FLOP fragment traffic vs one-tile-per-wave. K-step 32 with
// v_wmma_f32_16x16x32_bf16. M and Nc must be multiples of 32 (true here).
// ---------------------------------------------------------------------------
__global__ void gemm_wmma_bf16(const unsigned short* __restrict__ A,
                               const unsigned short* __restrict__ Bt,
                               const float* __restrict__ bias,
                               float* __restrict__ Cf,
                               unsigned short* __restrict__ Cbf,
                               int K, int Nc, int relu, int ntiles) {
    int wave = blockIdx.x * (blockDim.x >> 5) + (threadIdx.x >> 5);
    if (wave >= ntiles) return;              // whole-wave exit, EXEC stays full
    int lane   = threadIdx.x & 31;
    int tilesN = Nc >> 5;                    // 32-wide column tiles
    int tm = wave / tilesN;
    int tn = wave - tm * tilesN;

    // A fragment: lane holds row (lane&15); lanes<16 cover K {0..7,16..23},
// lanes>=16 cover K {8..15,24..31} per k-step.  B fragment (Bt row-major
// over [col,K]): same K split as A.
    const unsigned short* arow0 = A  + (size_t)(tm * 32 + (lane & 15)) * (size_t)K;
    const unsigned short* arow1 = arow0 + (size_t)16 * (size_t)K;
    const unsigned short* brow0 = Bt + (size_t)(tn * 32 + (lane & 15)) * (size_t)K;
    const unsigned short* brow1 = brow0 + (size_t)16 * (size_t)K;
    int aoff = (lane < 16) ? 0 : 8;
    int boff = (lane < 16) ? 0 : 8;

    v8f acc00 = {}, acc01 = {}, acc10 = {}, acc11 = {};
    for (int kk = 0; kk < K; kk += 32) {
        Frag a0, a1, b0, b1;
        a0.q[0] = *(const v4u*)(arow0 + kk + aoff);
        a0.q[1] = *(const v4u*)(arow0 + kk + 16 + aoff);
        a1.q[0] = *(const v4u*)(arow1 + kk + aoff);
        a1.q[1] = *(const v4u*)(arow1 + kk + 16 + aoff);
        b0.q[0] = *(const v4u*)(brow0 + kk + boff);
        b0.q[1] = *(const v4u*)(brow0 + kk + 16 + boff);
        b1.q[0] = *(const v4u*)(brow1 + kk + boff);
        b1.q[1] = *(const v4u*)(brow1 + kk + 16 + boff);
        acc00 = __builtin_amdgcn_wmma_f32_16x16x32_bf16(
            false, a0.v, false, b0.v, (short)0, acc00, false, false);
        acc01 = __builtin_amdgcn_wmma_f32_16x16x32_bf16(
            false, a0.v, false, b1.v, (short)0, acc01, false, false);
        acc10 = __builtin_amdgcn_wmma_f32_16x16x32_bf16(
            false, a1.v, false, b0.v, (short)0, acc10, false, false);
        acc11 = __builtin_amdgcn_wmma_f32_16x16x32_bf16(
            false, a1.v, false, b1.v, (short)0, acc11, false, false);
    }

    // C/D layout: VGPR j -> row (j + (lane<16 ? 0 : 8)), col (lane&15)
    int nb0 = tn * 32 + (lane & 15);
    int nb1 = nb0 + 16;
    int mb0 = tm * 32 + ((lane < 16) ? 0 : 8);
    int mb1 = mb0 + 16;
    float bv0 = bias ? bias[nb0] : 0.0f;
    float bv1 = bias ? bias[nb1] : 0.0f;
#pragma unroll
    for (int j = 0; j < 8; ++j) {
        float o00 = acc00[j] + bv0;
        float o01 = acc01[j] + bv1;
        float o10 = acc10[j] + bv0;
        float o11 = acc11[j] + bv1;
        if (relu) {
            o00 = fmaxf(o00, 0.0f); o01 = fmaxf(o01, 0.0f);
            o10 = fmaxf(o10, 0.0f); o11 = fmaxf(o11, 0.0f);
        }
        size_t r0 = (size_t)(mb0 + j) * (size_t)Nc;
        size_t r1 = (size_t)(mb1 + j) * (size_t)Nc;
        if (Cf) {
            Cf[r0 + nb0] = o00; Cf[r0 + nb1] = o01;
            Cf[r1 + nb0] = o10; Cf[r1 + nb1] = o11;
        }
        if (Cbf) {
            Cbf[r0 + nb0] = f2bf(o00); Cbf[r0 + nb1] = f2bf(o01);
            Cbf[r1 + nb0] = f2bf(o10); Cbf[r1 + nb1] = f2bf(o11);
        }
    }
}

// ---------------------------------------------------------------------------
// Elementwise helpers
// ---------------------------------------------------------------------------
__global__ void cvt_f32_bf16(const float* __restrict__ in,
                             unsigned short* __restrict__ out, int n) {
    int i = blockIdx.x * blockDim.x + threadIdx.x;
    if (i < n) out[i] = f2bf(in[i]);
}

// W[K,Nc] f32 -> out[Nc,K] bf16 (transpose for GEMM B operand)
__global__ void cvt_transpose(const float* __restrict__ W,
                              unsigned short* __restrict__ out, int K, int Nc) {
    int i = blockIdx.x * blockDim.x + threadIdx.x;
    if (i >= K * Nc) return;
    int kk = i / Nc, n = i - kk * Nc;
    out[(size_t)n * K + kk] = f2bf(W[i]);
}

__global__ void fill_f32(float* p, float v, int n) {
    int i = blockIdx.x * blockDim.x + threadIdx.x;
    if (i < n) p[i] = v;
}
__global__ void fill_u32(unsigned int* p, unsigned int v, int n) {
    int i = blockIdx.x * blockDim.x + threadIdx.x;
    if (i < n) p[i] = v;
}

// ---------------------------------------------------------------------------
// Edge attention: one wave per edge; 8 heads x 32 dims per edge.
// ---------------------------------------------------------------------------
__global__ void edge_logits(const float* __restrict__ q, const float* __restrict__ k,
                            const float* __restrict__ x,
                            const int* __restrict__ src, const int* __restrict__ dst,
                            float* __restrict__ alog) {
    int e = blockIdx.x * (blockDim.x >> 5) + (threadIdx.x >> 5);
    if (e >= EE) return;
    int lane = threadIdx.x & 31;
    const float* qr = q + (size_t)dst[e] * HH;
    const float* kr = k + (size_t)src[e] * HH;
    const float* xr = x + (size_t)e * HH;
#pragma unroll
    for (int c = 0; c < NH; ++c) {
        int idx = c * DH + lane;
        float p = qr[idx] * (kr[idx] + xr[idx]);
        for (int m = 16; m >= 1; m >>= 1) p += __shfl_xor(p, m, 32);
        if (lane == 0) alog[(size_t)e * NH + c] = p * INV_SQRT_DH;
    }
}

__device__ __forceinline__ unsigned int f32_enc(float f) {
    unsigned int u = __float_as_uint(f);
    return (u & 0x80000000u) ? ~u : (u | 0x80000000u);
}
__device__ __forceinline__ float f32_dec(unsigned int v) {
    unsigned int u = (v & 0x80000000u) ? (v & 0x7FFFFFFFu) : ~v;
    return __uint_as_float(u);
}

__global__ void edge_amax(const float* __restrict__ alog, const int* __restrict__ dst,
                          unsigned int* __restrict__ amax) {
    int i = blockIdx.x * blockDim.x + threadIdx.x;
    if (i >= EE * NH) return;
    int e = i >> 3, h = i & 7;
    atomicMax(&amax[(size_t)dst[e] * NH + h], f32_enc(alog[i]));
}

__global__ void edge_exp(const float* __restrict__ alog, const int* __restrict__ dst,
                         const unsigned int* __restrict__ amax,
                         float* __restrict__ ex, float* __restrict__ denom) {
    int i = blockIdx.x * blockDim.x + threadIdx.x;
    if (i >= EE * NH) return;
    int e = i >> 3, h = i & 7;
    float m = f32_dec(amax[(size_t)dst[e] * NH + h]);
    float v = expf(alog[i] - m);
    ex[i] = v;
    atomicAdd(&denom[(size_t)dst[e] * NH + h], v);
}

// scatter alpha*(v[src]+x) into hnew (hnew pre-loaded with h@Wh0+bh0)
__global__ void edge_agg(const float* __restrict__ vmat, const float* __restrict__ x,
                         const float* __restrict__ ex, const float* __restrict__ denom,
                         const int* __restrict__ src, const int* __restrict__ dst,
                         float* __restrict__ hnew) {
    int e = blockIdx.x * (blockDim.x >> 5) + (threadIdx.x >> 5);
    if (e >= EE) return;
    int lane = threadIdx.x & 31;
    int s = src[e], d = dst[e];
    const float* vr = vmat + (size_t)s * HH;
    const float* xr = x + (size_t)e * HH;
#pragma unroll
    for (int c = 0; c < NH; ++c) {
        float alpha = ex[(size_t)e * NH + c] / denom[(size_t)d * NH + c];
        int idx = c * DH + lane;
        atomicAdd(&hnew[(size_t)d * HH + idx], alpha * (vr[idx] + xr[idx]));
    }
}

// ---------------------------------------------------------------------------
// LayerNorm over H=256: one block (256 threads) per node. in2 optional add.
// ---------------------------------------------------------------------------
__global__ void layernorm(const float* __restrict__ in1, const float* __restrict__ in2,
                          const float* __restrict__ g, const float* __restrict__ b,
                          float* __restrict__ outf, unsigned short* __restrict__ outbf) {
    __shared__ float sm[2 * HH];
    int node = blockIdx.x, tid = threadIdx.x;
    float v = in1[(size_t)node * HH + tid];
    if (in2) v += in2[(size_t)node * HH + tid];
    sm[tid] = v;
    sm[HH + tid] = v * v;
    __syncthreads();
    for (int st = HH / 2; st > 0; st >>= 1) {
        if (tid < st) {
            sm[tid] += sm[tid + st];
            sm[HH + tid] += sm[HH + tid + st];
        }
        __syncthreads();
    }
    float mu  = sm[0] * (1.0f / HH);
    float var = sm[HH] * (1.0f / HH) - mu * mu;
    float y = (v - mu) * rsqrtf(var + LN_EPS) * g[tid] + b[tid];
    size_t off = (size_t)node * HH + tid;
    outf[off] = y;
    if (outbf) outbf[off] = f2bf(y);
}

// ---------------------------------------------------------------------------
// Readout
// ---------------------------------------------------------------------------
__global__ void seg_sum(const float* __restrict__ h, const int* __restrict__ gid,
                        float* __restrict__ hg) {
    int i = blockIdx.x * blockDim.x + threadIdx.x;
    if (i >= NN * HH) return;
    int node = i >> 8, c = i & 255;
    atomicAdd(&hg[(size_t)gid[node] * HH + c], h[i]);
}

__global__ void proj(const float* __restrict__ hg, const float* __restrict__ Wp,
                     float* __restrict__ out) {
    int t = threadIdx.x;            // 128 threads: (graph, out-col)
    int g = t >> 1, o = t & 1;
    float s = 0.0f;
    for (int c = 0; c < HH; ++c) s += hg[(size_t)g * HH + c] * Wp[c * 2 + o];
    out[t] = s;
}

// ---------------------------------------------------------------------------
static inline int cdiv(long long a, long long b) { return (int)((a + b - 1) / b); }

extern "C" void kernel_launch(void* const* d_in, const int* in_sizes, int n_in,
                              void* d_out, int out_size, void* d_ws, size_t ws_size,
                              hipStream_t stream) {
    const float* node_feat = (const float*)d_in[0];
    const float* edge_feat = (const float*)d_in[1];
    const int*   src       = (const int*)d_in[2];
    const int*   dst       = (const int*)d_in[3];
    const int*   gids      = (const int*)d_in[4];
    const float* W_node    = (const float*)d_in[5];
    const float* W_edge    = (const float*)d_in[6];
    const float* We        = (const float*)d_in[7];
    const float* be        = (const float*)d_in[8];
    const float* Wq        = (const float*)d_in[9];
    const float* bq        = (const float*)d_in[10];
    const float* Wk        = (const float*)d_in[11];
    const float* bk        = (const float*)d_in[12];
    const float* Wv        = (const float*)d_in[13];
    const float* bv        = (const float*)d_in[14];
    const float* Wh0       = (const float*)d_in[15];
    const float* bh0       = (const float*)d_in[16];
    const float* ln_a_g    = (const float*)d_in[17];
    const float* ln_a_b    = (const float*)d_in[18];
    const float* W1        = (const float*)d_in[19];
    const float* b1        = (const float*)d_in[20];
    const float* W2        = (const float*)d_in[21];
    const float* b2        = (const float*)d_in[22];
    const float* ln_l_g    = (const float*)d_in[23];
    const float* ln_l_b    = (const float*)d_in[24];
    const float* W_proj    = (const float*)d_in[25];
    float* out = (float*)d_out;

    // bump allocator over workspace
    size_t off = 0;
    auto alloc = [&](size_t bytes) -> char* {
        char* p = (char*)d_ws + off;
        off += (bytes + 255) & ~(size_t)255;
        return p;
    };
    unsigned short* nf_bf  = (unsigned short*)alloc((size_t)NN * AF * 2);
    unsigned short* ef_bf  = (unsigned short*)alloc((size_t)EE * BF * 2);
    unsigned short* e_bf   = (unsigned short*)alloc((size_t)EE * HH * 2);
    float*          x      = (float*)alloc((size_t)EE * HH * 4);
    float*          h      = (float*)alloc((size_t)NN * HH * 4);
    unsigned short* h_bf   = (unsigned short*)alloc((size_t)NN * HH * 2);
    float*          q      = (float*)alloc((size_t)NN * HH * 4);
    float*          k      = (float*)alloc((size_t)NN * HH * 4);
    float*          v      = (float*)alloc((size_t)NN * HH * 4);
    float*          hnew   = (float*)alloc((size_t)NN * HH * 4);
    float*          h1     = (float*)alloc((size_t)NN * HH * 4);
    unsigned short* h1_bf  = (unsigned short*)alloc((size_t)NN * HH * 2);
    unsigned short* t_bf   = (unsigned short*)alloc((size_t)NN * MH * 2);
    float*          h2     = (float*)alloc((size_t)NN * HH * 4);
    float*          alog   = (float*)alloc((size_t)EE * NH * 4);
    float*          exb    = (float*)alloc((size_t)EE * NH * 4);
    unsigned int*   amax   = (unsigned int*)alloc((size_t)NN * NH * 4);
    float*          denom  = (float*)alloc((size_t)NN * NH * 4);
    float*          hg     = (float*)alloc((size_t)GG * HH * 4);
    unsigned short* wbuf   = (unsigned short*)alloc((size_t)HH * MH * 2);
    (void)ws_size; (void)in_sizes; (void)n_in; (void)out_size;

    const int B = 256;
    auto gemm = [&](const unsigned short* A, const unsigned short* Bt, const float* bias,
                    float* Cf, unsigned short* Cbf, int M, int K, int Nc, int relu) {
        int ntiles = (M / 32) * (Nc / 32);
        gemm_wmma_bf16<<<cdiv(ntiles, 8), B, 0, stream>>>(A, Bt, bias, Cf, Cbf, K, Nc, relu, ntiles);
    };

    // --- embeddings -------------------------------------------------------
    cvt_f32_bf16<<<cdiv((size_t)NN * AF, B), B, 0, stream>>>(node_feat, nf_bf, NN * AF);
    cvt_f32_bf16<<<cdiv((size_t)EE * BF, B), B, 0, stream>>>(edge_feat, ef_bf, EE * BF);
    cvt_transpose<<<cdiv(AF * HH, B), B, 0, stream>>>(W_node, wbuf, AF, HH);
    gemm(nf_bf, wbuf, nullptr, h, h_bf, NN, AF, HH, 0);
    cvt_transpose<<<cdiv(BF * HH, B), B, 0, stream>>>(W_edge, wbuf, BF, HH);
    gemm(ef_bf, wbuf, nullptr, nullptr, e_bf, EE, BF, HH, 0);

    for (int i = 0; i < LL; ++i) {
        // edge projection x = e_ij @ We + be
        cvt_transpose<<<cdiv(HH * HH, B), B, 0, stream>>>(We + (size_t)i * HH * HH, wbuf, HH, HH);
        gemm(e_bf, wbuf, be + (size_t)i * HH, x, nullptr, EE, HH, HH, 0);
        // q, k, v, h@Wh0+bh0
        cvt_transpose<<<cdiv(HH * HH, B), B, 0, stream>>>(Wq + (size_t)i * HH * HH, wbuf, HH, HH);
        gemm(h_bf, wbuf, bq + (size_t)i * HH, q, nullptr, NN, HH, HH, 0);
        cvt_transpose<<<cdiv(HH * HH, B), B, 0, stream>>>(Wk + (size_t)i * HH * HH, wbuf, HH, HH);
        gemm(h_bf, wbuf, bk + (size_t)i * HH, k, nullptr, NN, HH, HH, 0);
        cvt_transpose<<<cdiv(HH * HH, B), B, 0, stream>>>(Wv + (size_t)i * HH * HH, wbuf, HH, HH);
        gemm(h_bf, wbuf, bv + (size_t)i * HH, v, nullptr, NN, HH, HH, 0);
        cvt_transpose<<<cdiv(HH * HH, B), B, 0, stream>>>(Wh0 + (size_t)i * HH * HH, wbuf, HH, HH);
        gemm(h_bf, wbuf, bh0 + (size_t)i * HH, hnew, nullptr, NN, HH, HH, 0);
        // edge softmax attention + aggregation (into hnew)
        fill_u32<<<cdiv(NN * NH, B), B, 0, stream>>>(amax, 0u, NN * NH);
        fill_f32<<<cdiv(NN * NH, B), B, 0, stream>>>(denom, 0.0f, NN * NH);
        edge_logits<<<cdiv(EE, 8), B, 0, stream>>>(q, k, x, src, dst, alog);
        edge_amax<<<cdiv((size_t)EE * NH, B), B, 0, stream>>>(alog, dst, amax);
        edge_exp<<<cdiv((size_t)EE * NH, B), B, 0, stream>>>(alog, dst, amax, exb, denom);
        edge_agg<<<cdiv(EE, 8), B, 0, stream>>>(v, x, exb, denom, src, dst, hnew);
        // transformer block
        layernorm<<<NN, HH, 0, stream>>>(hnew, nullptr, ln_a_g + (size_t)i * HH,
                                         ln_a_b + (size_t)i * HH, h1, h1_bf);
        cvt_transpose<<<cdiv(HH * MH, B), B, 0, stream>>>(W1 + (size_t)i * HH * MH, wbuf, HH, MH);
        gemm(h1_bf, wbuf, b1 + (size_t)i * MH, nullptr, t_bf, NN, HH, MH, 1);
        cvt_transpose<<<cdiv(MH * HH, B), B, 0, stream>>>(W2 + (size_t)i * MH * HH, wbuf, MH, HH);
        gemm(t_bf, wbuf, b2 + (size_t)i * HH, h2, nullptr, NN, MH, HH, 0);
        layernorm<<<NN, HH, 0, stream>>>(h1, h2, ln_l_g + (size_t)i * HH,
                                         ln_l_b + (size_t)i * HH, h, h_bf);
    }

    // --- readout ----------------------------------------------------------
    fill_f32<<<cdiv(GG * HH, B), B, 0, stream>>>(hg, 0.0f, GG * HH);
    seg_sum<<<cdiv((size_t)NN * HH, B), B, 0, stream>>>(h, gids, hg);
    proj<<<1, 128, 0, stream>>>(hg, W_proj, out);
}
